// Block_395136991964
// MI455X (gfx1250) — compile-verified, matrix operand order checked by reading, not run on hardware
//
#include <hip/hip_runtime.h>

// ---------------------------------------------------------------------------
// Transformer block (LN1 -> QKV -> RoPE -> flash attn -> proj(+res) -> LN2 ->
// SwiGLU MLP(+res)) for MI455X / gfx1250, wave32, bf16 WMMA with f32 accum.
// GEMM tiles are staged into LDS via the Tensor Data Mover (double-buffered,
// TENSORcnt-tracked) so each 128x32 operand tile is fetched from L2 once per
// block instead of once per wave.
// ---------------------------------------------------------------------------

#define Bsz   8
#define Ntok  1374
#define Npad  1376            // next multiple of 32
#define Cdim  768
#define Hn    12
#define HD    64
#define HID   3072
#define C3    (3 * Cdim)      // 2304
#define Mrows (Bsz * Ntok)    // 10992
#define SCALE 0.125f          // 1/sqrt(64)

typedef unsigned short u16;
typedef unsigned int   u32;
typedef unsigned long long u64;
typedef __attribute__((ext_vector_type(16))) __bf16 v16bf;
typedef __attribute__((ext_vector_type(8)))  float  v8f;
typedef __attribute__((ext_vector_type(4)))  unsigned v4u;
typedef __attribute__((ext_vector_type(8)))  int      v8i;
typedef __attribute__((ext_vector_type(4)))  int      v4i;

union FragBF { v16bf v; uint4 u[2]; };

static __device__ __forceinline__ u16 f2bf(float f) {
  union { float f; u32 u; } a; a.f = f;
  u32 u = a.u;
  u += 0x7fffu + ((u >> 16) & 1u);   // round-to-nearest-even
  return (u16)(u >> 16);
}

static __device__ __forceinline__ v8f wmma_bf16(v16bf a, v16bf b, v8f c) {
  // D = A(16x32 bf16) * B(32x16 bf16) + C(16x16 f32)
  return __builtin_amdgcn_wmma_f32_16x16x32_bf16(false, a, false, b,
                                                 (short)0, c, false, false);
}

// ---------------------------------------------------------------------------
// TDM: load a 2-D tile (32 elems wide x tile_rows tall, row stride =
// stride_elems, element = 2 bytes) from global into LDS at lds_byte.
// D# per CDNA5 ISA §8.3/8.4: group0 = {count|flags, lds_addr, global_addr,
// type=2}; group1 carries data_size / tensor dims / tile dims / dim0 stride.
// tensor_dim1 = tensor_rows gives zero-fill for the M-edge rows.
// ---------------------------------------------------------------------------
#if __has_builtin(__builtin_amdgcn_tensor_load_to_lds)
#define USE_TDM 1
#else
#define USE_TDM 0
#endif

#if USE_TDM
static __device__ __forceinline__ void tdm_load_2d(unsigned lds_byte,
                                                   const u16* gptr,
                                                   int stride_elems,
                                                   int tensor_rows) {
  u64 ga = (u64)(uintptr_t)gptr;
  v4u g0;
  g0.x = 1u;                                        // count=1, user descriptor
  g0.y = lds_byte;                                  // lds_addr [63:32]
  g0.z = (u32)(ga & 0xffffffffu);                   // global_addr lo
  g0.w = (u32)((ga >> 32) & 0x01ffffffu) | (2u << 30);  // addr hi | type=2

  const u32 td0 = 32u;                      // tensor_dim0 (elements)
  const u32 td1 = (u32)tensor_rows;         // tensor_dim1 (rows, OOB->0)
  const u32 tl0 = 32u, tl1 = 128u;          // tile dims
  const u64 s0  = (u64)(u32)stride_elems;   // tensor_dim0_stride

  v8i g1;
  g1[0] = (int)(1u << 16);                               // data_size=1 (2B)
  g1[1] = (int)((td0 & 0xffffu) << 16);                  // td0[15:0] @ [63:48]
  g1[2] = (int)((td0 >> 16) | ((td1 & 0xffffu) << 16));  // td0 hi | td1 lo
  g1[3] = (int)((td1 >> 16) | (tl0 << 16));              // td1 hi | tile_dim0
  g1[4] = (int)tl1;                                      // tile_dim1 (dim2=0)
  g1[5] = (int)(s0 & 0xffffffffu);                       // stride0 [191:160]
  g1[6] = (int)((s0 >> 32) & 0xffffu);                   // stride0 hi
  g1[7] = 0;

  const v4i gz = {0, 0, 0, 0};
#if defined(__clang_major__) && (__clang_major__ >= 23)
  const v8i gz8 = {0, 0, 0, 0, 0, 0, 0, 0};
  __builtin_amdgcn_tensor_load_to_lds(g0, g1, gz, gz, gz8, 0);
#else
  __builtin_amdgcn_tensor_load_to_lds(g0, g1, gz, gz, 0);
#endif
}
#endif

// ---------------------------------------------------------------------------
// fp32 -> bf16 elementwise (weights)
// ---------------------------------------------------------------------------
__global__ __launch_bounds__(256) void cvt_kernel(const float* __restrict__ in,
                                                  u16* __restrict__ out, int n) {
  int i = blockIdx.x * 256 + threadIdx.x;
  if (i < n) out[i] = f2bf(in[i]);
}

// ---------------------------------------------------------------------------
// LayerNorm over C=768, fp32 in -> bf16 out.  One block (256 thr) per token.
// ---------------------------------------------------------------------------
__global__ __launch_bounds__(256) void ln_kernel(const float* __restrict__ x,
                                                 const float* __restrict__ g,
                                                 const float* __restrict__ b,
                                                 u16* __restrict__ out) {
  int row  = blockIdx.x;
  int tid  = threadIdx.x;
  int lane = tid & 31, wid = tid >> 5;
  const float* xr = x + (size_t)row * Cdim;

  float v[3], s = 0.f, ss = 0.f;
#pragma unroll
  for (int i = 0; i < 3; ++i) {
    v[i] = xr[tid + i * 256];
    s += v[i]; ss += v[i] * v[i];
  }
#pragma unroll
  for (int m = 1; m <= 16; m <<= 1) {
    s  += __shfl_xor(s,  m, 32);
    ss += __shfl_xor(ss, m, 32);
  }
  __shared__ float rs[8], rss[8];
  if (lane == 0) { rs[wid] = s; rss[wid] = ss; }
  __syncthreads();
  float S = 0.f, SS = 0.f;
#pragma unroll
  for (int w = 0; w < 8; ++w) { S += rs[w]; SS += rss[w]; }
  float mean = S * (1.f / Cdim);
  float var  = SS * (1.f / Cdim) - mean * mean;
  float rstd = rsqrtf(var + 1e-5f);
#pragma unroll
  for (int i = 0; i < 3; ++i) {
    int c = tid + i * 256;
    out[(size_t)row * Cdim + c] = f2bf((v[i] - mean) * rstd * g[c] + b[c]);
  }
}

// ---------------------------------------------------------------------------
// Generic GEMM: Out[M,Nout] = A_bf16[M,K] * W_bf16[Nout,K]^T  (+ epilogue)
//   MODE 0: out f32 = acc + bias
//   MODE 1: out f32 = acc + bias + aux        (residual add)
//   MODE 2: out bf16 = silu(aux) * (acc+bias) (SwiGLU gate; aux = x1 f32)
// Block = 128 thr (4 waves, 2x2), wave tile 64x64, K step 32.
// Operand tiles (128x32 A and B) staged into LDS by the TDM, double-buffered:
// wave 0 issues tensor_load_to_lds for tile k+1, all waves compute on tile k
// from LDS (ds_load_b128 fragments), then s_wait_tensorcnt + barrier.
// ---------------------------------------------------------------------------
template <int MODE>
__global__ __launch_bounds__(128) void gemm_kernel(
    const u16* __restrict__ A, const u16* __restrict__ W,
    const float* __restrict__ bias, const float* __restrict__ aux,
    void* __restrict__ outv, int Mr, int Nout, int K) {
  __shared__ u16 smA[2][128 * 32];
  __shared__ u16 smB[2][128 * 32];

  int lane = threadIdx.x & 31;
  int half = lane >> 4;
  int r    = lane & 15;
  int wave = threadIdx.x >> 5;
  int wm = wave >> 1, wn = wave & 1;
  int rowB = blockIdx.x * 128;
  int colB = blockIdx.y * 128;
  int row0 = rowB + wm * 64;
  int col0 = colB + wn * 64;

  const v8f vz = {0.f, 0.f, 0.f, 0.f, 0.f, 0.f, 0.f, 0.f};
  v8f acc[4][4];
#pragma unroll
  for (int i = 0; i < 4; ++i)
#pragma unroll
    for (int j = 0; j < 4; ++j) acc[i][j] = vz;

  const u16* Abase = A + (size_t)rowB * K;
  const u16* Wbase = W + (size_t)colB * K;
  int remRows = Mr - rowB;
  if (remRows > 128) remRows = 128;

#if USE_TDM
  if (wave == 0) {
    tdm_load_2d((u32)(uintptr_t)&smA[0][0], Abase, K, remRows);
    tdm_load_2d((u32)(uintptr_t)&smB[0][0], Wbase, K, 128);
    __builtin_amdgcn_s_wait_tensorcnt(0);
  }
  __syncthreads();
#endif

  int cb = 0;
  for (int k0 = 0; k0 < K; k0 += 32) {
#if USE_TDM
    int kn = k0 + 32;
    if (kn < K && wave == 0) {
      tdm_load_2d((u32)(uintptr_t)&smA[cb ^ 1][0], Abase + kn, K, remRows);
      tdm_load_2d((u32)(uintptr_t)&smB[cb ^ 1][0], Wbase + kn, K, 128);
    }
#else
    {  // cooperative fallback staging: thread t copies row t of both tiles
      int t = threadIdx.x;
      int ra = t; if (ra >= remRows) ra = remRows - 1;
      const uint4* ga = (const uint4*)(Abase + (size_t)ra * K + k0);
      const uint4* gw = (const uint4*)(Wbase + (size_t)t * K + k0);
      uint4* la = (uint4*)&smA[0][t * 32];
      uint4* lw = (uint4*)&smB[0][t * 32];
#pragma unroll
      for (int q = 0; q < 4; ++q) { la[q] = ga[q]; lw[q] = gw[q]; }
    }
    __syncthreads();
#endif

    FragBF a[4], w[4];
#pragma unroll
    for (int mi = 0; mi < 4; ++mi) {
      const u16* p = &smA[cb][(wm * 64 + mi * 16 + r) * 32 + half * 8];
      a[mi].u[0] = *(const uint4*)p;
      a[mi].u[1] = *(const uint4*)(p + 16);
    }
#pragma unroll
    for (int ni = 0; ni < 4; ++ni) {
      const u16* p = &smB[cb][(wn * 64 + ni * 16 + r) * 32 + half * 8];
      w[ni].u[0] = *(const uint4*)p;
      w[ni].u[1] = *(const uint4*)(p + 16);
    }
#pragma unroll
    for (int mi = 0; mi < 4; ++mi)
#pragma unroll
      for (int ni = 0; ni < 4; ++ni)
        acc[mi][ni] = wmma_bf16(a[mi].v, w[ni].v, acc[mi][ni]);

#if USE_TDM
    if (kn < K && wave == 0) __builtin_amdgcn_s_wait_tensorcnt(0);
    __syncthreads();
    cb ^= 1;
#else
    __syncthreads();
#endif
  }

#pragma unroll
  for (int mi = 0; mi < 4; ++mi) {
#pragma unroll
    for (int ni = 0; ni < 4; ++ni) {
      int col = col0 + ni * 16 + r;
      float bv = bias[col];
#pragma unroll
      for (int g = 0; g < 8; ++g) {
        int row = row0 + mi * 16 + g + 8 * half;   // C/D VGPR layout
        if (row < Mr) {
          size_t idx = (size_t)row * Nout + col;
          float v = acc[mi][ni][g] + bv;
          if (MODE == 0) {
            ((float*)outv)[idx] = v;
          } else if (MODE == 1) {
            ((float*)outv)[idx] = v + aux[idx];
          } else {                                  // MODE 2
            float x1 = aux[idx];
            float sl = x1 / (1.f + __expf(-x1));
            ((u16*)outv)[idx] = f2bf(sl * v);
          }
        }
      }
    }
  }
}

// ---------------------------------------------------------------------------
// RoPE + QKV split: qkv f32 [B,N,3,H,HD] -> q/k rotated bf16 [B,H,Npad,HD],
// v transposed bf16 [B,H,HD,Npad] (so P*V B-fragments are contiguous).
// Pad rows are zeroed; attention masks cols >= N anyway.
// ---------------------------------------------------------------------------
__global__ __launch_bounds__(256) void rope_kernel(
    const float* __restrict__ qkv, const float* __restrict__ freqs,
    u16* __restrict__ qb, u16* __restrict__ kb, u16* __restrict__ vt) {
  long long idx = (long long)blockIdx.x * 256 + threadIdx.x;
  const long long total = (long long)Bsz * Hn * Npad * HD;
  if (idx >= total) return;
  int d = idx & 63;
  int n = (int)((idx >> 6) % Npad);
  int h = (int)((idx >> 6) / Npad % Hn);
  int b = (int)(idx / ((long long)64 * Npad * Hn));

  size_t qk_i = ((size_t)(b * Hn + h) * Npad + n) * HD + d;
  size_t vt_i = ((size_t)(b * Hn + h) * HD + d) * Npad + n;
  if (n < Ntok) {
    size_t base = ((size_t)(b * Ntok + n)) * 3 * Cdim;
    float q  = qkv[base + 0 * Cdim + h * HD + d];
    float k  = qkv[base + 1 * Cdim + h * HD + d];
    float v  = qkv[base + 2 * Cdim + h * HD + d];
    int   d2 = (d < 32) ? d + 32 : d - 32;
    float sg = (d < 32) ? -1.f : 1.f;
    float q2 = qkv[base + 0 * Cdim + h * HD + d2];
    float k2 = qkv[base + 1 * Cdim + h * HD + d2];
    float f  = freqs[(size_t)n * HD + d];
    float cf = __cosf(f), sf = __sinf(f);
    qb[qk_i] = f2bf(q * cf + sg * q2 * sf);
    kb[qk_i] = f2bf(k * cf + sg * k2 * sf);
    vt[vt_i] = f2bf(v);
  } else {
    qb[qk_i] = 0; kb[qk_i] = 0; vt[vt_i] = 0;
  }
}

// ---------------------------------------------------------------------------
// Flash attention: one wave per (b, h, 16-row Q tile). Key tiles of 32.
//   S = Q K^T : 2 WMMA per 16-col sub-tile (K-dim 64 = 2x32 chunks)
//   online softmax (cross-lane max/sum via shfl_xor within 16-lane halves)
//   P staged through LDS (same-wave DS ops are in-order) to re-layout the
//   C-layout accumulator into an A-layout bf16 fragment, then O += P V.
// Output written as bf16 [M, C] ready for the proj GEMM.
// ---------------------------------------------------------------------------
__global__ __launch_bounds__(32) void flash_kernel(
    const u16* __restrict__ qb, const u16* __restrict__ kb,
    const u16* __restrict__ vt, u16* __restrict__ ob) {
  int lane = threadIdx.x;
  int half = lane >> 4;
  int r    = lane & 15;
  int q0 = blockIdx.x * 16;
  int h  = blockIdx.y;
  int b  = blockIdx.z;
  int bh = b * Hn + h;

  const v8f vz = {0.f, 0.f, 0.f, 0.f, 0.f, 0.f, 0.f, 0.f};
  FragBF qa[2];
#pragma unroll
  for (int kc = 0; kc < 2; ++kc) {
    const u16* p = qb + ((size_t)(bh * Npad + q0 + r)) * HD + kc * 32 + half * 8;
    qa[kc].u[0] = *(const uint4*)p;
    qa[kc].u[1] = *(const uint4*)(p + 16);
  }

  float m_[8], l_[8];
#pragma unroll
  for (int g = 0; g < 8; ++g) { m_[g] = -3e38f; l_[g] = 0.f; }
  v8f oa[4];
#pragma unroll
  for (int ni = 0; ni < 4; ++ni) oa[ni] = vz;

  __shared__ u16 pls[16 * 32];

  for (int j0 = 0; j0 < Npad; j0 += 32) {
    v8f s[2] = {vz, vz};
#pragma unroll
    for (int ns = 0; ns < 2; ++ns) {
      const u16* kr = kb + ((size_t)(bh * Npad + j0 + ns * 16 + r)) * HD + half * 8;
      FragBF f0, f1;
      f0.u[0] = *(const uint4*)kr;        f0.u[1] = *(const uint4*)(kr + 16);
      f1.u[0] = *(const uint4*)(kr + 32); f1.u[1] = *(const uint4*)(kr + 48);
      s[ns] = wmma_bf16(qa[0].v, f0.v, s[ns]);
      s[ns] = wmma_bf16(qa[1].v, f1.v, s[ns]);
    }
    bool ok0 = (j0 + r)      < Ntok;
    bool ok1 = (j0 + 16 + r) < Ntok;
#pragma unroll
    for (int g = 0; g < 8; ++g) {
      float v0 = ok0 ? s[0][g] * SCALE : -3e38f;
      float v1 = ok1 ? s[1][g] * SCALE : -3e38f;
      float mx = fmaxf(v0, v1);
#pragma unroll
      for (int msk = 1; msk <= 8; msk <<= 1) mx = fmaxf(mx, __shfl_xor(mx, msk, 32));
      float mn = fmaxf(m_[g], mx);
      float al = __expf(m_[g] - mn);
      float p0 = __expf(v0 - mn), p1 = __expf(v1 - mn);
      float rs = p0 + p1;
#pragma unroll
      for (int msk = 1; msk <= 8; msk <<= 1) rs += __shfl_xor(rs, msk, 32);
      m_[g] = mn;
      l_[g] = l_[g] * al + rs;
#pragma unroll
      for (int ni = 0; ni < 4; ++ni) oa[ni][g] *= al;
      int row16 = g + 8 * half;
      pls[row16 * 32 + r]      = f2bf(p0);
      pls[row16 * 32 + 16 + r] = f2bf(p1);
    }
    // same-wave LDS store->load is in order (DScnt); reload as A fragment
    FragBF pf;
    pf.u[0] = *(const uint4*)(&pls[r * 32 + half * 8]);
    pf.u[1] = *(const uint4*)(&pls[r * 32 + half * 8 + 16]);
#pragma unroll
    for (int ni = 0; ni < 4; ++ni) {
      const u16* vr = vt + ((size_t)(bh * HD + ni * 16 + r)) * Npad + j0 + half * 8;
      FragBF vf;
      vf.u[0] = *(const uint4*)vr;
      vf.u[1] = *(const uint4*)(vr + 16);
      oa[ni] = wmma_bf16(pf.v, vf.v, oa[ni]);
    }
  }

#pragma unroll
  for (int ni = 0; ni < 4; ++ni) {
#pragma unroll
    for (int g = 0; g < 8; ++g) {
      int row16 = g + 8 * half;
      int n = q0 + row16;
      if (n < Ntok) {
        float inv = 1.f / l_[g];
        ob[((size_t)(b * Ntok + n)) * Cdim + h * HD + ni * 16 + r] =
            f2bf(oa[ni][g] * inv);
      }
    }
  }
}

// ---------------------------------------------------------------------------
// Host-side orchestration
// ---------------------------------------------------------------------------
extern "C" void kernel_launch(void* const* d_in, const int* in_sizes, int n_in,
                              void* d_out, int out_size, void* d_ws, size_t ws_size,
                              hipStream_t stream) {
  const float* x      = (const float*)d_in[0];
  const float* freqs  = (const float*)d_in[1];
  const float* ln1_g  = (const float*)d_in[2];
  const float* ln1_b  = (const float*)d_in[3];
  const float* qkv_w  = (const float*)d_in[4];
  const float* qkv_b  = (const float*)d_in[5];
  const float* proj_w = (const float*)d_in[6];
  const float* proj_b = (const float*)d_in[7];
  const float* ln2_g  = (const float*)d_in[8];
  const float* ln2_b  = (const float*)d_in[9];
  const float* w1     = (const float*)d_in[10];
  const float* b1     = (const float*)d_in[11];
  const float* w2     = (const float*)d_in[12];
  const float* b2     = (const float*)d_in[13];
  const float* w3     = (const float*)d_in[14];
  const float* b3     = (const float*)d_in[15];
  float* out = (float*)d_out;

  // workspace carve-out
  char* ws = (char*)d_ws;
  size_t off = 0;
  auto take = [&](size_t bytes) -> char* {
    char* p = ws + off;
    off = (off + bytes + 255) & ~(size_t)255;
    return p;
  };
  const size_t NWQ = (size_t)C3 * Cdim, NWP = (size_t)Cdim * Cdim;
  const size_t NW1 = (size_t)HID * Cdim, NW3 = (size_t)Cdim * HID;
  const size_t NQK = (size_t)Bsz * Hn * Npad * HD;

  u16*   wqb  = (u16*)take(NWQ * 2);
  u16*   wpb  = (u16*)take(NWP * 2);
  u16*   w1b  = (u16*)take(NW1 * 2);
  u16*   w2b  = (u16*)take(NW1 * 2);
  u16*   w3b  = (u16*)take(NW3 * 2);
  u16*   hb   = (u16*)take((size_t)Mrows * Cdim * 2);   // LN1 out, reused for LN2
  float* bufA = (float*)take((size_t)Mrows * HID * 4);  // qkv f32, then x1 f32
  u16*   qb   = (u16*)take(NQK * 2);
  u16*   kb   = (u16*)take(NQK * 2);
  u16*   vtb  = (u16*)take(NQK * 2);
  u16*   ob   = (u16*)take((size_t)Mrows * Cdim * 2);
  float* xres = (float*)take((size_t)Mrows * Cdim * 4);
  u16*   hidb = (u16*)take((size_t)Mrows * HID * 2);
  float* qkvf = bufA;
  float* x1f  = bufA;

  // 1) weights -> bf16
  auto cvt = [&](const float* src, u16* dst, size_t n) {
    cvt_kernel<<<dim3((unsigned)((n + 255) / 256)), 256, 0, stream>>>(src, dst, (int)n);
  };
  cvt(qkv_w, wqb, NWQ);
  cvt(proj_w, wpb, NWP);
  cvt(w1, w1b, NW1);
  cvt(w2, w2b, NW1);
  cvt(w3, w3b, NW3);

  const dim3 gB(128);
  const unsigned MT = (Mrows + 127) / 128;   // 86

  // 2) LN1 -> bf16
  ln_kernel<<<dim3(Mrows), 256, 0, stream>>>(x, ln1_g, ln1_b, hb);
  // 3) QKV GEMM (f32 out)
  gemm_kernel<0><<<dim3(MT, C3 / 128), gB, 0, stream>>>(hb, wqb, qkv_b, nullptr,
                                                        qkvf, Mrows, C3, Cdim);
  // 4) RoPE split
  {
    long long tot = (long long)Bsz * Hn * Npad * HD;
    rope_kernel<<<dim3((unsigned)((tot + 255) / 256)), 256, 0, stream>>>(
        qkvf, freqs, qb, kb, vtb);
  }
  // 5) flash attention
  flash_kernel<<<dim3(Npad / 16, Hn, Bsz), 32, 0, stream>>>(qb, kb, vtb, ob);
  // 6) proj GEMM + residual (x)
  gemm_kernel<1><<<dim3(MT, Cdim / 128), gB, 0, stream>>>(ob, wpb, proj_b, x,
                                                          xres, Mrows, Cdim, Cdim);
  // 7) LN2 -> bf16
  ln_kernel<<<dim3(Mrows), 256, 0, stream>>>(xres, ln2_g, ln2_b, hb);
  // 8) x1 = h2 @ w1^T + b1 (f32)
  gemm_kernel<0><<<dim3(MT, HID / 128), gB, 0, stream>>>(hb, w1b, b1, nullptr,
                                                         x1f, Mrows, HID, Cdim);
  // 9) hid = silu(x1) * (h2 @ w2^T + b2)  (bf16)
  gemm_kernel<2><<<dim3(MT, HID / 128), gB, 0, stream>>>(hb, w2b, b2, x1f,
                                                         hidb, Mrows, HID, Cdim);
  // 10) out = xres + hid @ w3^T + b3
  gemm_kernel<1><<<dim3(MT, Cdim / 128), gB, 0, stream>>>(hidb, w3b, b3, xres,
                                                          out, Mrows, Cdim, HID);
}
